// ModularTemporalCell_86887188398402
// MI455X (gfx1250) — matrix operand order repaired and checked by reading, not compiled, not run on hardware
//
#include <hip/hip_runtime.h>
#include <math.h>

// ---------------------------------------------------------------------------
// MI455X (gfx1250) implementation. wave32, WMMA 16x16x32 bf16 (f32 accum).
// GEMMs tiled 128x128, K-step 32, 8 waves/block, each wave owns 32x64
// (2x4 WMMA tiles). Activations are produced in bf16 by the elementwise
// kernels (halves activation traffic; A tiles stage as b128 copies).
// Weights are fp32 in global (read once) and converted to bf16 while staging.
// 2-stage register pipeline overlaps next-tile global loads with WMMA, plus
// global_prefetch two tiles ahead on the streaming weights. The SSM (WB/WC)
// GEMM is a fused dual-accumulator kernel whose epilogue computes
// h_memory_new and y_mem (cross-lane reduce over n) in-register.
// ---------------------------------------------------------------------------

typedef __bf16 v16bf __attribute__((ext_vector_type(16)));
typedef float  v8f   __attribute__((ext_vector_type(8)));
typedef unsigned short u16;

union Frag { v16bf v; uint4 q[2]; };

#define DEV static __device__ __forceinline__

constexpr int Bsz = 4096;   // batch
constexpr int Dd  = 1024;   // model dim
constexpr int Nst = 16;     // SSM state dim
constexpr size_t BD = (size_t)Bsz * Dd;

constexpr int LDK = 40;     // LDS row stride in bf16 elems (80B, 16B aligned)

DEV u16 f2bf(float f) {                        // round-to-nearest-even fp32->bf16
  unsigned int u = __float_as_uint(f);
  u += 0x7FFFu + ((u >> 16) & 1u);
  return (u16)(u >> 16);
}
DEV float sigm(float x) { return 1.f / (1.f + __expf(-x)); }

enum { ACT_NONE = 0, ACT_SIG = 1, ACT_TANH = 2, ACT_SP01 = 3 };

// ---------------------------------------------------------------------------
// C = act(A_bf16[M,K] @ W_f32[K,N] + bias) ; C fp32.
// M,N multiples of 128; K multiple of 32.
// ---------------------------------------------------------------------------
template <int ACT>
__global__ __launch_bounds__(256) void gemm_wmma_kernel(
    const u16* __restrict__ A, const float* __restrict__ W,
    const float* __restrict__ bias, float* __restrict__ C,
    int M, int Ncols, int K) {
  __shared__ u16 As[128 * LDK];
  __shared__ u16 Bs[128 * LDK];

  const int tid  = threadIdx.x;
  const int wid  = tid >> 5;
  const int lane = tid & 31;
  const int half = lane >> 4;
  const int ln   = lane & 15;
  const int m0   = blockIdx.y * 128;
  const int n0   = blockIdx.x * 128;
  const int wrow = (wid >> 1) * 32;
  const int wcol = (wid & 1) * 64;

  v8f acc[2][4];
#pragma unroll
  for (int a = 0; a < 2; ++a)
#pragma unroll
    for (int b = 0; b < 4; ++b)
#pragma unroll
      for (int i = 0; i < 8; ++i) acc[a][b][i] = 0.f;

  uint4  rA[2];      // A tile: 128x32 bf16, 2x(uint4=8 bf16) per thread
  float4 rW[4];      // W tile: 32x128 f32, 4x(float4) per thread

  // ---- prologue loads (k = 0) -------------------------------------------
#pragma unroll
  for (int p = 0; p < 2; ++p) {
    int idx = p * 256 + tid;                 // 0..511, row = idx/4, q = idx%4
    rA[p] = *(const uint4*)(A + (size_t)(m0 + (idx >> 2)) * K + ((idx & 3) << 3));
  }
#pragma unroll
  for (int p = 0; p < 4; ++p) {
    int idx = p * 256 + tid;                 // kr = idx/32, nq = (idx%32)*4
    rW[p] = *(const float4*)(W + (size_t)(idx >> 5) * Ncols + n0 + ((idx & 31) << 2));
  }

  for (int k0 = 0; k0 < K; k0 += 32) {
    __syncthreads();
    // ---- commit staged registers to LDS ---------------------------------
#pragma unroll
    for (int p = 0; p < 2; ++p) {
      int idx = p * 256 + tid;
      *(uint4*)(&As[(idx >> 2) * LDK + ((idx & 3) << 3)]) = rA[p];
    }
#pragma unroll
    for (int p = 0; p < 4; ++p) {
      int idx = p * 256 + tid;
      int kr  = idx >> 5;
      int nq  = (idx & 31) << 2;
      Bs[(nq + 0) * LDK + kr] = f2bf(rW[p].x);
      Bs[(nq + 1) * LDK + kr] = f2bf(rW[p].y);
      Bs[(nq + 2) * LDK + kr] = f2bf(rW[p].z);
      Bs[(nq + 3) * LDK + kr] = f2bf(rW[p].w);
    }
    __syncthreads();

    // ---- pipeline: load k0+32 tile while WMMAs execute ------------------
    const int kn = (k0 + 32 < K) ? (k0 + 32) : k0;   // clamp (redundant last)
#pragma unroll
    for (int p = 0; p < 2; ++p) {
      int idx = p * 256 + tid;
      rA[p] = *(const uint4*)(A + (size_t)(m0 + (idx >> 2)) * K + kn + ((idx & 3) << 3));
    }
#pragma unroll
    for (int p = 0; p < 4; ++p) {
      int idx = p * 256 + tid;
      rW[p] = *(const float4*)(W + (size_t)(kn + (idx >> 5)) * Ncols + n0 + ((idx & 31) << 2));
    }
    if (k0 + 64 < K)                                  // stream hint 2 tiles out
      __builtin_prefetch(W + (size_t)(k0 + 64 + (tid >> 5)) * Ncols + n0 + ((tid & 31) << 2), 0, 1);

    // A frag: lanes 0-15 hold K 0..15; lanes 16-31 hold K 16..31
    Frag af[2];
    const int ka0 = half ? 16 : 0;
#pragma unroll
    for (int mt = 0; mt < 2; ++mt) {
      int r = wrow + mt * 16 + ln;
      af[mt].q[0] = *(const uint4*)(&As[r * LDK + ka0]);
      af[mt].q[1] = *(const uint4*)(&As[r * LDK + ka0 + 8]);
    }
    // B frag: lanes 0-15 hold K 0..15 of col n; lanes 16-31 hold K 16..31
    const int kb = half ? 16 : 0;
#pragma unroll
    for (int nt = 0; nt < 4; ++nt) {
      Frag bf;
      int c = wcol + nt * 16 + ln;
      bf.q[0] = *(const uint4*)(&Bs[c * LDK + kb]);
      bf.q[1] = *(const uint4*)(&Bs[c * LDK + kb + 8]);
#pragma unroll
      for (int mt = 0; mt < 2; ++mt)
        acc[mt][nt] = __builtin_amdgcn_wmma_f32_16x16x32_bf16(
            false, af[mt].v, false, bf.v, (short)0, acc[mt][nt], false, false);
    }
  }

  // epilogue: VGPR i -> rows (i, i+8) across half-waves, col = lane&15
#pragma unroll
  for (int mt = 0; mt < 2; ++mt)
#pragma unroll
    for (int nt = 0; nt < 4; ++nt) {
      int col = n0 + wcol + nt * 16 + ln;
      float bia = bias[col];
#pragma unroll
      for (int i = 0; i < 8; ++i) {
        int row = m0 + wrow + mt * 16 + half * 8 + i;
        float v = acc[mt][nt][i] + bia;
        if (ACT == ACT_SIG)  v = sigm(v);
        if (ACT == ACT_TANH) v = tanhf(v);
        if (ACT == ACT_SP01) v = 0.1f * ((v > 20.f) ? v : log1pf(__expf(v)));
        C[(size_t)row * Ncols + col] = v;
      }
    }
}

// ---------------------------------------------------------------------------
// Fused SSM GEMM: Bm = gm@WB+bB and Cm = gm@WC+bC for one 128x128 tile
// (cols j = d*16+n), dual accumulators sharing the A tile; epilogue applies
//   h_new = alpha*exp(delta*(-exp(A_log)))*h_mem + beta*(delta*Bm*x)
//   y_mem[b,d] = sum_n Cm*h_new      (shfl_xor reduce, n == lane&15)
// ---------------------------------------------------------------------------
__global__ __launch_bounds__(256) void ssm_gemm_kernel(
    const u16* __restrict__ A, const float* __restrict__ WB,
    const float* __restrict__ bB, const float* __restrict__ WC,
    const float* __restrict__ bC, const float* __restrict__ alpha,
    const float* __restrict__ beta, const float* __restrict__ delta,
    const float* __restrict__ x, const float* __restrict__ A_log,
    const float* __restrict__ h_mem, float* __restrict__ h_new_out,
    float* __restrict__ y_mem, int K) {
  constexpr int NC = Dd * Nst;  // 16384 columns
  __shared__ u16 As [128 * LDK];
  __shared__ u16 BsB[128 * LDK];
  __shared__ u16 BsC[128 * LDK];

  const int tid  = threadIdx.x;
  const int wid  = tid >> 5;
  const int lane = tid & 31;
  const int half = lane >> 4;
  const int ln   = lane & 15;
  const int m0   = blockIdx.y * 128;
  const int n0   = blockIdx.x * 128;
  const int wrow = (wid >> 1) * 32;
  const int wcol = (wid & 1) * 64;

  v8f accB[2][4], accC[2][4];
#pragma unroll
  for (int a = 0; a < 2; ++a)
#pragma unroll
    for (int b = 0; b < 4; ++b)
#pragma unroll
      for (int i = 0; i < 8; ++i) { accB[a][b][i] = 0.f; accC[a][b][i] = 0.f; }

  uint4  rA[2];
  float4 rWB[4], rWC[4];

#pragma unroll
  for (int p = 0; p < 2; ++p) {
    int idx = p * 256 + tid;
    rA[p] = *(const uint4*)(A + (size_t)(m0 + (idx >> 2)) * K + ((idx & 3) << 3));
  }
#pragma unroll
  for (int p = 0; p < 4; ++p) {
    int idx = p * 256 + tid;
    size_t off = (size_t)(idx >> 5) * NC + n0 + ((idx & 31) << 2);
    rWB[p] = *(const float4*)(WB + off);
    rWC[p] = *(const float4*)(WC + off);
  }

  for (int k0 = 0; k0 < K; k0 += 32) {
    __syncthreads();
#pragma unroll
    for (int p = 0; p < 2; ++p) {
      int idx = p * 256 + tid;
      *(uint4*)(&As[(idx >> 2) * LDK + ((idx & 3) << 3)]) = rA[p];
    }
#pragma unroll
    for (int p = 0; p < 4; ++p) {
      int idx = p * 256 + tid;
      int kr  = idx >> 5;
      int nq  = (idx & 31) << 2;
      BsB[(nq + 0) * LDK + kr] = f2bf(rWB[p].x);
      BsB[(nq + 1) * LDK + kr] = f2bf(rWB[p].y);
      BsB[(nq + 2) * LDK + kr] = f2bf(rWB[p].z);
      BsB[(nq + 3) * LDK + kr] = f2bf(rWB[p].w);
      BsC[(nq + 0) * LDK + kr] = f2bf(rWC[p].x);
      BsC[(nq + 1) * LDK + kr] = f2bf(rWC[p].y);
      BsC[(nq + 2) * LDK + kr] = f2bf(rWC[p].z);
      BsC[(nq + 3) * LDK + kr] = f2bf(rWC[p].w);
    }
    __syncthreads();

    const int kn = (k0 + 32 < K) ? (k0 + 32) : k0;
#pragma unroll
    for (int p = 0; p < 2; ++p) {
      int idx = p * 256 + tid;
      rA[p] = *(const uint4*)(A + (size_t)(m0 + (idx >> 2)) * K + kn + ((idx & 3) << 3));
    }
#pragma unroll
    for (int p = 0; p < 4; ++p) {
      int idx = p * 256 + tid;
      size_t off = (size_t)(kn + (idx >> 5)) * NC + n0 + ((idx & 31) << 2);
      rWB[p] = *(const float4*)(WB + off);
      rWC[p] = *(const float4*)(WC + off);
    }
    if (k0 + 64 < K) {
      size_t off = (size_t)(k0 + 64 + (tid >> 5)) * NC + n0 + ((tid & 31) << 2);
      __builtin_prefetch(WB + off, 0, 1);
      __builtin_prefetch(WC + off, 0, 1);
    }

    Frag af[2];
    const int ka0 = half ? 16 : 0;
#pragma unroll
    for (int mt = 0; mt < 2; ++mt) {
      int r = wrow + mt * 16 + ln;
      af[mt].q[0] = *(const uint4*)(&As[r * LDK + ka0]);
      af[mt].q[1] = *(const uint4*)(&As[r * LDK + ka0 + 8]);
    }
    const int kb = half ? 16 : 0;
#pragma unroll
    for (int nt = 0; nt < 4; ++nt) {
      int c = wcol + nt * 16 + ln;
      Frag fb, fc;
      fb.q[0] = *(const uint4*)(&BsB[c * LDK + kb]);
      fb.q[1] = *(const uint4*)(&BsB[c * LDK + kb + 8]);
      fc.q[0] = *(const uint4*)(&BsC[c * LDK + kb]);
      fc.q[1] = *(const uint4*)(&BsC[c * LDK + kb + 8]);
#pragma unroll
      for (int mt = 0; mt < 2; ++mt) {
        accB[mt][nt] = __builtin_amdgcn_wmma_f32_16x16x32_bf16(
            false, af[mt].v, false, fb.v, (short)0, accB[mt][nt], false, false);
        accC[mt][nt] = __builtin_amdgcn_wmma_f32_16x16x32_bf16(
            false, af[mt].v, false, fc.v, (short)0, accC[mt][nt], false, false);
      }
    }
  }

  // SSM epilogue: each 16-wide WMMA column tile is one d, n == ln
#pragma unroll
  for (int mt = 0; mt < 2; ++mt)
#pragma unroll
    for (int nt = 0; nt < 4; ++nt) {
      int colbase = n0 + wcol + nt * 16;
      int d   = colbase >> 4;
      int col = colbase + ln;
      float bBc = bB[col], bCc = bC[col];
      float negA = -__expf(A_log[col]);
#pragma unroll
      for (int i = 0; i < 8; ++i) {
        int b = m0 + wrow + mt * 16 + half * 8 + i;
        size_t bd = (size_t)b * Dd + d;
        float Bm = accB[mt][nt][i] + bBc;
        float Cm = accC[mt][nt][i] + bCc;
        float al = alpha[bd], be = beta[bd], dl = delta[bd], xv = x[bd];
        float Abar = __expf(dl * negA);
        float hnew = al * (Abar * h_mem[bd * Nst + ln]) + be * (dl * Bm * xv);
        h_new_out[bd * Nst + ln] = hnew;
        float v = Cm * hnew;
        v += __shfl_xor(v, 1);
        v += __shfl_xor(v, 2);
        v += __shfl_xor(v, 4);
        v += __shfl_xor(v, 8);
        if (ln == 0) y_mem[bd] = v;
      }
    }
}

// ---------------------------------------------------------------------------
// Block-per-row elementwise / LayerNorm kernels (256 threads, 4 cols/thread)
// ---------------------------------------------------------------------------
DEV void blockReduce2(float& s1, float& s2, float* sh) {
#pragma unroll
  for (int m = 16; m >= 1; m >>= 1) { s1 += __shfl_xor(s1, m); s2 += __shfl_xor(s2, m); }
  int wid = threadIdx.x >> 5, lane = threadIdx.x & 31;
  if (lane == 0) { sh[wid] = s1; sh[8 + wid] = s2; }
  __syncthreads();
  if (threadIdx.x == 0) {
    float a = 0.f, b = 0.f;
    for (int i = 0; i < 8; ++i) { a += sh[i]; b += sh[8 + i]; }
    sh[16] = a; sh[17] = b;
  }
  __syncthreads();
  s1 = sh[16]; s2 = sh[17];
}

// gi = bf16[h_logic, LN(x)] ; gm = bf16[mean_n(h_memory), x]
__global__ __launch_bounds__(256) void prep_kernel(
    const float* __restrict__ x, const float* __restrict__ h_logic,
    const float* __restrict__ h_mem, const float* __restrict__ g,
    const float* __restrict__ lb, u16* __restrict__ gi, u16* __restrict__ gm) {
  __shared__ float sh[32];
  int b = blockIdx.x, t = threadIdx.x;
  const float* xr = x + (size_t)b * Dd;
  float xv[4], s = 0.f, s2 = 0.f;
#pragma unroll
  for (int i = 0; i < 4; ++i) { xv[i] = xr[t * 4 + i]; s += xv[i]; s2 += xv[i] * xv[i]; }
  blockReduce2(s, s2, sh);
  float mean = s * (1.f / Dd);
  float inv  = rsqrtf(s2 * (1.f / Dd) - mean * mean + 1e-5f);
  size_t rb = (size_t)b * 2 * Dd;
#pragma unroll
  for (int i = 0; i < 4; ++i) {
    int c = t * 4 + i;
    gi[rb + Dd + c] = f2bf((xv[i] - mean) * inv * g[c] + lb[c]);
    gi[rb + c]      = f2bf(h_logic[(size_t)b * Dd + c]);
    const float* hm = h_mem + ((size_t)b * Dd + c) * Nst;
    float hs = 0.f;
#pragma unroll
    for (int n = 0; n < Nst; ++n) hs += hm[n];
    gm[rb + c]      = f2bf(hs * (1.f / Nst));
    gm[rb + Dd + c] = f2bf(xv[i]);
  }
}

// gc = bf16[r*h_logic, x_norm(copied from gi upper half)]
__global__ __launch_bounds__(256) void gc_kernel(
    const float* __restrict__ r, const float* __restrict__ h_logic,
    const u16* __restrict__ gi, u16* __restrict__ gc) {
  size_t i = (size_t)blockIdx.x * blockDim.x + threadIdx.x;
  if (i >= BD) return;
  size_t b = i / Dd, d = i % Dd;
  gc[b * 2 * Dd + d]      = f2bf(r[i] * h_logic[i]);
  gc[b * 2 * Dd + Dd + d] = gi[b * 2 * Dd + Dd + d];
}

// h_new = (1-z)h + z*h_cand -> d_out ; y_logic = LN(h_new)  (fp32)
__global__ __launch_bounds__(256) void logic_fin_kernel(
    const float* __restrict__ z, const float* __restrict__ hc,
    const float* __restrict__ h_logic, const float* __restrict__ g,
    const float* __restrict__ lb, float* __restrict__ hn_out,
    float* __restrict__ y_logic) {
  __shared__ float sh[32];
  int b = blockIdx.x, t = threadIdx.x;
  size_t base = (size_t)b * Dd;
  float v[4], s = 0.f, s2 = 0.f;
#pragma unroll
  for (int i = 0; i < 4; ++i) {
    int c = t * 4 + i;
    float zz = z[base + c], h = h_logic[base + c];
    float hv = (1.f - zz) * h + zz * hc[base + c];
    v[i] = hv; hn_out[base + c] = hv; s += hv; s2 += hv * hv;
  }
  blockReduce2(s, s2, sh);
  float mean = s * (1.f / Dd);
  float inv  = rsqrtf(s2 * (1.f / Dd) - mean * mean + 1e-5f);
#pragma unroll
  for (int i = 0; i < 4; ++i) {
    int c = t * 4 + i;
    y_logic[base + c] = (v[i] - mean) * inv * g[c] + lb[c];
  }
}

// y_memory = LN(y_mem + Dp*x) (fp32) ; ri = bf16[x, y_logic, y_memory]
__global__ __launch_bounds__(256) void mem_ln_kernel(
    const float* __restrict__ y_mem, const float* __restrict__ x,
    const float* __restrict__ Dp, const float* __restrict__ g,
    const float* __restrict__ lb, const float* __restrict__ y_logic,
    float* __restrict__ y_memory, u16* __restrict__ ri) {
  __shared__ float sh[32];
  int b = blockIdx.x, t = threadIdx.x;
  size_t base = (size_t)b * Dd;
  float v[4], s = 0.f, s2 = 0.f;
#pragma unroll
  for (int i = 0; i < 4; ++i) {
    int c = t * 4 + i;
    float vv = y_mem[base + c] + Dp[c] * x[base + c];
    v[i] = vv; s += vv; s2 += vv * vv;
  }
  blockReduce2(s, s2, sh);
  float mean = s * (1.f / Dd);
  float inv  = rsqrtf(s2 * (1.f / Dd) - mean * mean + 1e-5f);
  size_t rb = (size_t)b * 3 * Dd;
#pragma unroll
  for (int i = 0; i < 4; ++i) {
    int c = t * 4 + i;
    float ym = (v[i] - mean) * inv * g[c] + lb[c];
    y_memory[base + c]  = ym;
    ri[rb + c]          = f2bf(x[base + c]);
    ri[rb + Dd + c]     = f2bf(y_logic[base + c]);
    ri[rb + 2 * Dd + c] = f2bf(ym);
  }
}

// rw = softmax(t@R2 + r2b) ; ymix = bf16(rw0*y_logic + rw1*y_memory)
__global__ __launch_bounds__(256) void route_kernel(
    const float* __restrict__ tt, const float* __restrict__ R2,
    const float* __restrict__ r2b, const float* __restrict__ y_logic,
    const float* __restrict__ y_memory, u16* __restrict__ ymix) {
  __shared__ float sh[32];
  int b = blockIdx.x, t = threadIdx.x;
  size_t base = (size_t)b * Dd;
  float d0 = 0.f, d1 = 0.f;
#pragma unroll
  for (int i = 0; i < 4; ++i) {
    int c = t * 4 + i;
    float tv = tt[base + c];
    d0 += tv * R2[c * 2 + 0];
    d1 += tv * R2[c * 2 + 1];
  }
  blockReduce2(d0, d1, sh);
  d0 += r2b[0]; d1 += r2b[1];
  float mx = fmaxf(d0, d1);
  float e0 = __expf(d0 - mx), e1 = __expf(d1 - mx);
  float w0 = e0 / (e0 + e1), w1 = e1 / (e0 + e1);
#pragma unroll
  for (int i = 0; i < 4; ++i) {
    int c = t * 4 + i;
    ymix[base + c] = f2bf(w0 * y_logic[base + c] + w1 * y_memory[base + c]);
  }
}

// y_out = LN(y_pre) (fp32) ; gb = bf16[y_out, x]
__global__ __launch_bounds__(256) void yout_kernel(
    const float* __restrict__ ypre, const float* __restrict__ x,
    const float* __restrict__ g, const float* __restrict__ lb,
    float* __restrict__ yout, u16* __restrict__ gb) {
  __shared__ float sh[32];
  int b = blockIdx.x, t = threadIdx.x;
  size_t base = (size_t)b * Dd;
  float v[4], s = 0.f, s2 = 0.f;
#pragma unroll
  for (int i = 0; i < 4; ++i) {
    int c = t * 4 + i;
    float vv = ypre[base + c];
    v[i] = vv; s += vv; s2 += vv * vv;
  }
  blockReduce2(s, s2, sh);
  float mean = s * (1.f / Dd);
  float inv  = rsqrtf(s2 * (1.f / Dd) - mean * mean + 1e-5f);
  size_t rb = (size_t)b * 2 * Dd;
#pragma unroll
  for (int i = 0; i < 4; ++i) {
    int c = t * 4 + i;
    float yo = (v[i] - mean) * inv * g[c] + lb[c];
    yout[base + c]  = yo;
    gb[rb + c]      = f2bf(yo);
    gb[rb + Dd + c] = f2bf(x[base + c]);
  }
}

// y = gamma * y_out
__global__ __launch_bounds__(256) void final_kernel(
    const float* __restrict__ gamma, const float* __restrict__ yout,
    float* __restrict__ y) {
  size_t i = (size_t)blockIdx.x * blockDim.x + threadIdx.x;
  if (i >= BD) return;
  y[i] = gamma[i] * yout[i];
}

// ---------------------------------------------------------------------------
// Launch
// ---------------------------------------------------------------------------
extern "C" void kernel_launch(void* const* d_in, const int* in_sizes, int n_in,
                              void* d_out, int out_size, void* d_ws, size_t ws_size,
                              hipStream_t stream) {
  const float* x        = (const float*)d_in[0];
  const float* h_logic  = (const float*)d_in[1];
  const float* h_memory = (const float*)d_in[2];
  const float* ln_pre_g = (const float*)d_in[3];
  const float* ln_pre_b = (const float*)d_in[4];
  const float* Wz = (const float*)d_in[5];
  const float* Wr = (const float*)d_in[6];
  const float* Wh = (const float*)d_in[7];
  const float* bz = (const float*)d_in[8];
  const float* br = (const float*)d_in[9];
  const float* bh = (const float*)d_in[10];
  const float* ln_logic_g = (const float*)d_in[11];
  const float* ln_logic_b = (const float*)d_in[12];
  const float* Wa = (const float*)d_in[13];
  const float* Wb = (const float*)d_in[14];
  const float* Wd = (const float*)d_in[15];
  const float* ba = (const float*)d_in[16];
  const float* bb = (const float*)d_in[17];
  const float* bd = (const float*)d_in[18];
  const float* A_log = (const float*)d_in[19];
  const float* WB = (const float*)d_in[20];
  const float* bB = (const float*)d_in[21];
  const float* WC = (const float*)d_in[22];
  const float* bC = (const float*)d_in[23];
  const float* Dp = (const float*)d_in[24];
  const float* ln_mem_g = (const float*)d_in[25];
  const float* ln_mem_b = (const float*)d_in[26];
  const float* R1  = (const float*)d_in[27];
  const float* r1b = (const float*)d_in[28];
  const float* R2  = (const float*)d_in[29];
  const float* r2b = (const float*)d_in[30];
  const float* Wout = (const float*)d_in[31];
  const float* bout = (const float*)d_in[32];
  const float* ln_fus_g = (const float*)d_in[33];
  const float* ln_fus_b = (const float*)d_in[34];
  const float* Wg = (const float*)d_in[35];
  const float* bg = (const float*)d_in[36];

  // workspace (float units). bf16 buffers occupy half their logical width.
  float* ws = (float*)d_ws;
  const size_t OF_GI = 0;              // bf16 [B,2D]  -> BD floats
  const size_t OF_GM = OF_GI + BD;     // bf16 [B,2D]
  const size_t OF_GC = OF_GM + BD;     // bf16 [B,2D]
  const size_t OF_Z  = OF_GC + BD;     // fp32 [B,D] each below
  const size_t OF_R  = OF_Z  + BD;
  const size_t OF_HC = OF_R  + BD;
  const size_t OF_AL = OF_HC + BD;
  const size_t OF_BE = OF_AL + BD;
  const size_t OF_DL = OF_BE + BD;
  const size_t OF_YM = OF_DL + BD;     // y_mem
  const size_t OF_YL = OF_YM + BD;     // y_logic
  const size_t OF_YY = OF_YL + BD;     // y_memory
  const size_t WS_FLOATS = OF_YY + BD; // 12*BD floats ~= 192 MB
  if (ws_size < WS_FLOATS * sizeof(float)) return;

  u16* giق  = nullptr; (void)giق; // (no-op; keep ASCII below)
  u16*   gi_u   = (u16*)(ws + OF_GI);
  u16*   gm_u   = (u16*)(ws + OF_GM);
  u16*   gc_u   = (u16*)(ws + OF_GC);
  // late-stage aliases (lifetimes disjoint):
  u16*   ri_u   = (u16*)(ws + OF_GM);  // bf16 [B,3D] spans GM + half GC (dead)
  float* tvec   = ws + OF_Z;
  u16*   ymix_u = (u16*)(ws + OF_R);
  float* ypre   = ws + OF_HC;
  float* yout   = ws + OF_AL;
  u16*   gb_u   = (u16*)(ws + OF_BE);  // bf16 [B,2D] fits in BE (BD floats)
  float* gamma  = ws + OF_YM;

  float* y_out_glob  = (float*)d_out;            // [B,D]
  float* h_logic_new = (float*)d_out + BD;       // [B,D]
  float* h_mem_new   = (float*)d_out + 2 * BD;   // [B,D,N]

  const dim3 blk(256);
  const dim3 ew((unsigned)((BD + 255) / 256));
  const dim3 rows((unsigned)Bsz);
  const dim3 gD(Dd / 128, Bsz / 128);            // N=1024 GEMMs
  const dim3 gDN(Dd * Nst / 128, Bsz / 128);     // N=16384 SSM GEMM

  // 1) prep: gi=[h_logic, LN(x)], gm=[mean(h_mem), x]   (bf16)
  prep_kernel<<<rows, blk, 0, stream>>>(x, h_logic, h_memory, ln_pre_g, ln_pre_b,
                                        gi_u, gm_u);
  // 2) logic gates
  gemm_wmma_kernel<ACT_SIG><<<gD, blk, 0, stream>>>(gi_u, Wz, bz, ws + OF_Z,
                                                    Bsz, Dd, 2 * Dd);
  gemm_wmma_kernel<ACT_SIG><<<gD, blk, 0, stream>>>(gi_u, Wr, br, ws + OF_R,
                                                    Bsz, Dd, 2 * Dd);
  // 3) memory gates
  gemm_wmma_kernel<ACT_SIG><<<gD, blk, 0, stream>>>(gm_u, Wa, ba, ws + OF_AL,
                                                    Bsz, Dd, 2 * Dd);
  gemm_wmma_kernel<ACT_SIG><<<gD, blk, 0, stream>>>(gm_u, Wb, bb, ws + OF_BE,
                                                    Bsz, Dd, 2 * Dd);
  gemm_wmma_kernel<ACT_SP01><<<gD, blk, 0, stream>>>(gm_u, Wd, bd, ws + OF_DL,
                                                     Bsz, Dd, 2 * Dd);
  // 4) candidate state
  gc_kernel<<<ew, blk, 0, stream>>>(ws + OF_R, h_logic, gi_u, gc_u);
  gemm_wmma_kernel<ACT_TANH><<<gD, blk, 0, stream>>>(gc_u, Wh, bh, ws + OF_HC,
                                                     Bsz, Dd, 2 * Dd);
  logic_fin_kernel<<<rows, blk, 0, stream>>>(ws + OF_Z, ws + OF_HC, h_logic,
                                             ln_logic_g, ln_logic_b,
                                             h_logic_new, ws + OF_YL);
  // 5) fused SSM GEMM (Bm & Cm + state update + y_mem reduction)
  ssm_gemm_kernel<<<gDN, blk, 0, stream>>>(gm_u, WB, bB, WC, bC,
                                           ws + OF_AL, ws + OF_BE, ws + OF_DL,
                                           x, A_log, h_memory,
                                           h_mem_new, ws + OF_YM, 2 * Dd);
  // 6) memory LN + fusion input
  mem_ln_kernel<<<rows, blk, 0, stream>>>(ws + OF_YM, x, Dp, ln_mem_g, ln_mem_b,
                                          ws + OF_YL, ws + OF_YY, ri_u);
  // 7) fusion
  gemm_wmma_kernel<ACT_TANH><<<gD, blk, 0, stream>>>(ri_u, R1, r1b, tvec,
                                                     Bsz, Dd, 3 * Dd);
  route_kernel<<<rows, blk, 0, stream>>>(tvec, R2, r2b, ws + OF_YL, ws + OF_YY,
                                         ymix_u);
  gemm_wmma_kernel<ACT_NONE><<<gD, blk, 0, stream>>>(ymix_u, Wout, bout, ypre,
                                                     Bsz, Dd, Dd);
  yout_kernel<<<rows, blk, 0, stream>>>(ypre, x, ln_fus_g, ln_fus_b, yout, gb_u);
  // 8) output gate
  gemm_wmma_kernel<ACT_SIG><<<gD, blk, 0, stream>>>(gb_u, Wg, bg, gamma,
                                                    Bsz, Dd, 2 * Dd);
  final_kernel<<<ew, blk, 0, stream>>>(gamma, yout, y_out_glob);
}
